// Encoder_42417097016185
// MI455X (gfx1250) — matrix operand order repaired and checked by reading, not compiled, not run on hardware
//
#include <hip/hip_runtime.h>
#include <hip/hip_bf16.h>

typedef __attribute__((ext_vector_type(16))) _Float16 v16h;
typedef __attribute__((ext_vector_type(8)))  _Float16 v8h;
typedef __attribute__((ext_vector_type(8)))  float    v8f;

#define BN_EPS 1e-5f
#define KIDS   32

__device__ __forceinline__ int refl(int p, int n) {
  p = (p < 0) ? -p : p;
  p = (p >= n) ? (2 * n - 2 - p) : p;
  return p;
}

// ---------------------------------------------------------------------------
// 7x7 reflect-pad conv, 3->32, fp32 NCHW in -> fp32 NHWC out. Weights in LDS.
// Geometry hardcoded to 512x512 (shifts only).
// ---------------------------------------------------------------------------
__global__ void conv0_reflect7(const float* __restrict__ x, const float* __restrict__ w,
                               const float* __restrict__ bias, float* __restrict__ out) {
  __shared__ float lw[3 * 49 * 32];          // [(ci*49+tap)*32 + co]
  for (int i = threadIdx.x; i < 3 * 49 * 32; i += blockDim.x) {
    int co = i & 31; int rest = i >> 5; int tap = rest % 49; int ci = rest / 49;
    lw[i] = w[(co * 3 + ci) * 49 + tap];     // w: (32,3,7,7) OIHW
  }
  __syncthreads();
  int pix = blockIdx.x * blockDim.x + threadIdx.x;
  int xq = pix & 511; int y = (pix >> 9) & 511; int b = pix >> 18;
  float acc[32];
#pragma unroll
  for (int co = 0; co < 32; ++co) acc[co] = bias[co];
  for (int ky = 0; ky < 7; ++ky) {
    int iy = refl(y - 3 + ky, 512);
    for (int kx = 0; kx < 7; ++kx) {
      int ix = refl(xq - 3 + kx, 512);
      int tap = ky * 7 + kx;
#pragma unroll
      for (int ci = 0; ci < 3; ++ci) {
        float v = x[((((size_t)(b * 3 + ci) << 9) + iy) << 9) + ix];
        const float* lp = &lw[(ci * 49 + tap) * 32];
#pragma unroll
        for (int co = 0; co < 32; ++co) acc[co] = fmaf(v, lp[co], acc[co]);
      }
    }
  }
  float* op = out + (size_t)pix * 32;
#pragma unroll
  for (int co = 0; co < 32; ++co) op[co] = acc[co];
}

// ---------------------------------------------------------------------------
// BatchNorm statistics: per-channel sum & sumsq over NHWC fp32 (C = pow2).
// ---------------------------------------------------------------------------
__global__ void bn_stats(const float* __restrict__ xin, float* __restrict__ stats,
                         int total, int C) {
  int t = threadIdx.x;
  int base = blockIdx.x * 16384;
  int c0 = (base + t) & (C - 1);
  int c1 = (base + t + 256) & (C - 1);
  float s0 = 0.f, q0 = 0.f, s1 = 0.f, q1 = 0.f;
  for (int k = 0; k < 64; ++k) {
    int i = base + t + 256 * k;
    if (i < total) {
      float v = xin[i];
      if (k & 1) { s1 += v; q1 += v * v; } else { s0 += v; q0 += v * v; }
    }
  }
  atomicAdd(&stats[c0], s0); atomicAdd(&stats[C + c0], q0);
  atomicAdd(&stats[c1], s1); atomicAdd(&stats[C + c1], q1);
}

// ---------------------------------------------------------------------------
// Zero the 1-pixel halo border of a padded NHWC f16 activation buffer.
// ---------------------------------------------------------------------------
__global__ void zero_halo(_Float16* __restrict__ act, int B, int H, int W, int C, int lgC) {
  int Wp = W + 2, Hp = H + 2;
  int perim = 2 * Wp + 2 * H;
  int total = B * perim * C;
  int idx = blockIdx.x * blockDim.x + threadIdx.x;
  if (idx >= total) return;
  int c = idx & (C - 1);
  int p = idx >> lgC;
  int b = p / perim; int pp = p % perim;
  int y, x;
  if (pp < Wp)            { y = 0;        x = pp; }
  else if (pp < 2 * Wp)   { y = Hp - 1;   x = pp - Wp; }
  else { int q = pp - 2 * Wp; y = 1 + (q >> 1); x = (q & 1) ? (Wp - 1) : 0; }
  act[(((size_t)(b * Hp + y) * Wp + x) << lgC) + c] = (_Float16)0.f;
}

// ---------------------------------------------------------------------------
// Fused BN(train-mode) + ReLU + quantize, writing into padded NHWC f16.
// All decomposition via shift/mask (pow2 dims).
// ---------------------------------------------------------------------------
__global__ void bn_relu_pad_f16(const float* __restrict__ xin, const float* __restrict__ stats,
                                const float* __restrict__ g, const float* __restrict__ bb,
                                _Float16* __restrict__ act, int total, int C, int lgC,
                                int lgW, int lgH, float invN) {
  int i = blockIdx.x * blockDim.x + threadIdx.x;
  if (i >= total) return;
  int c  = i & (C - 1);
  int p  = i >> lgC;
  int xx = p & ((1 << lgW) - 1);
  int p2 = p >> lgW;
  int yy = p2 & ((1 << lgH) - 1);
  int b  = p2 >> lgH;
  float m = stats[c] * invN;
  float v = stats[C + c] * invN - m * m;
  float inv = rsqrtf(v + BN_EPS);
  float y = (xin[i] - m) * inv * g[c] + bb[c];
  y = y > 0.f ? y : 0.f;
  int Wp = (1 << lgW) + 2, Hp = (1 << lgH) + 2;
  act[(((size_t)(b * Hp + yy + 1) * Wp + xx + 1) << lgC) + c] = (_Float16)y;
}

// ---------------------------------------------------------------------------
// Weight repack into WMMA B-fragment-native order (ISA 7.12.2).
// ---------------------------------------------------------------------------
__global__ void repack_w_down(const float* __restrict__ w, _Float16* __restrict__ dst,
                              int Cin, int Cout) {
  int K = 9 * Cin;
  int idx = blockIdx.x * blockDim.x + threadIdx.x;
  if (idx >= K * Cout) return;
  int n = idx % Cout; int k = idx / Cout;
  int tap = k / Cin; int ci = k % Cin;
  float v = w[(size_t)(n * Cin + ci) * 9 + tap];      // (cout,cin,3,3), tap=ky*3+kx
  int kc = k >> 5, kl = k & 31, hi = (kl >> 3) & 1, h = (kl & 7) | ((kl >> 4) << 3);
  int nt = n >> 4, lane = hi * 16 + (n & 15);
  int Nt = Cout >> 4;
  dst[((size_t)(kc * Nt + nt) * 32 + lane) * 16 + h] = (_Float16)v;
}

__global__ void repack_w_up(const float* __restrict__ w, _Float16* __restrict__ dst,
                            int Cin, int Cout, int ntaps, int4 tky, int4 tkx) {
  int K = ntaps * Cin;
  int idx = blockIdx.x * blockDim.x + threadIdx.x;
  if (idx >= K * Cout) return;
  int n = idx % Cout; int k = idx / Cout;
  int tap = k / Cin; int ci = k % Cin;
  int kys[4] = {tky.x, tky.y, tky.z, tky.w};
  int kxs[4] = {tkx.x, tkx.y, tkx.z, tkx.w};
  float v = w[((size_t)(n * Cin + ci) * 3 + kys[tap]) * 3 + kxs[tap]];
  int kc = k >> 5, kl = k & 31, hi = (kl >> 3) & 1, h = (kl & 7) | ((kl >> 4) << 3);
  int nt = n >> 4, lane = hi * 16 + (n & 15);
  int Nt = Cout >> 4;
  dst[((size_t)(kc * Nt + nt) * 32 + lane) * 16 + h] = (_Float16)v;
}

// ---------------------------------------------------------------------------
// Stride-2 3x3 conv as implicit GEMM on v_wmma_f32_16x16x32_f16.
// Zero-halo padded input => unconditional A loads. One wave computes
// 16 pixels x (16*NPW) output channels: each A fragment feeds NPW (up to 4)
// independent WMMA chains, filling the WMMA hazard/co-exec slots.
// ---------------------------------------------------------------------------
template <int Cin, int Cout>
__global__ void conv_down_wmma(const _Float16* __restrict__ in, const _Float16* __restrict__ wf,
                               const float* __restrict__ bias, float* __restrict__ out,
                               int Hin, int Win, int lgMt, int lgHout, int tiles) {
  constexpr int CC    = Cin / 32;
  constexpr int Nt    = Cout / 16;
  constexpr int NPW   = (Nt >= 4) ? 4 : Nt;     // N-tiles per wave
  constexpr int NtG   = Nt / NPW;               // wave groups along N (pow2)
  constexpr int LgNtG = __builtin_ctz(NtG);
  int Hp = Hin + 2, Wp = Win + 2;
  int Hout = Hin >> 1, Wout = Win >> 1;
  int wid = (blockIdx.x * blockDim.x + threadIdx.x) >> 5;
  int lane = threadIdx.x & 31;
  if (wid >= tiles) return;                     // whole-wave exit: EXEC stays all-ones
  int np = wid & (NtG - 1);
  int t  = wid >> LgNtG;
  int mx = t & ((1 << lgMt) - 1);  t >>= lgMt;
  int oy = t & ((1 << lgHout) - 1);
  int b  = t >> lgHout;
  int lr = lane & 15, hi = lane >> 4;
  int ox = (mx << 4) + lr;
  v8f acc[NPW];
#pragma unroll
  for (int j = 0; j < NPW; ++j) acc[j] = (v8f){};
  const _Float16* wbase = wf + ((size_t)(np * NPW) * 32 + lane) * 16;   // (kc=0, nt=NPW*np)
#pragma unroll
  for (int ky = 0; ky < 3; ++ky) {
    int iy = oy * 2 + ky;                       // padded coordinate, always in-bounds
    const _Float16* rowp = in + (size_t)(b * Hp + iy) * Wp * Cin;
#pragma unroll
    for (int kx = 0; kx < 3; ++kx) {
      int ix = ox * 2 + kx;                     // padded coordinate
      const _Float16* ap = rowp + (size_t)ix * Cin + hi * 8;
      const _Float16* bp = wbase + (size_t)(ky * 3 + kx) * CC * Nt * 512;
#pragma unroll 2
      for (int cc = 0; cc < CC; ++cc) {
        v8h alo = *(const v8h*)(ap);            // halves 0..7 : K = hi*8..+7
        v8h ahi = *(const v8h*)(ap + 16);       // halves 8..15: K = 16+hi*8..+7
        v16h a = __builtin_shufflevector(alo, ahi, 0,1,2,3,4,5,6,7,8,9,10,11,12,13,14,15);
        v16h bfr[NPW];
#pragma unroll
        for (int j = 0; j < NPW; ++j) bfr[j] = *(const v16h*)(bp + j * 512);
#pragma unroll
        for (int j = 0; j < NPW; ++j)
          acc[j] = __builtin_amdgcn_wmma_f32_16x16x32_f16(false, a, false, bfr[j], (short)0,
                                                          acc[j], false, false);
        ap += 32;
        bp += (size_t)Nt * 512;
      }
    }
  }
  int n0 = (np * NPW) * 16 + lr;
  float* outb = out + ((size_t)(b * Hout + oy) * Wout + (mx << 4)) * Cout;
#pragma unroll
  for (int j = 0; j < NPW; ++j) {
    float bv = bias[n0 + 16 * j];
#pragma unroll
    for (int r = 0; r < 8; ++r) {
      int m = (hi << 3) + r;                    // C/D layout: lanes16-31 hold M=8..15
      outb[(size_t)m * Cout + n0 + 16 * j] = acc[j][r] + bv;
    }
  }
}

// ---------------------------------------------------------------------------
// ConvTranspose(3x3,s2,p1,op1) as 4 parity sub-convs on padded input.
// Parity P is a template parameter -> tap tables fold to constants.
// ---------------------------------------------------------------------------
template <int Cin, int Cout, int P>
__global__ void conv_up_wmma(const _Float16* __restrict__ in, const _Float16* __restrict__ wf,
                             const float* __restrict__ bias, float* __restrict__ out,
                             int Hin, int Win, int lgMt, int lgHin, int tiles) {
  constexpr int PY = P >> 1, PX = P & 1;
  constexpr int NTAPS = (P == 0) ? 1 : (P == 3) ? 4 : 2;
  constexpr int DY[4][4] = {{0,0,0,0},{0,0,0,0},{0,1,0,0},{0,0,1,1}};
  constexpr int DX[4][4] = {{0,0,0,0},{0,1,0,0},{0,0,0,0},{0,1,0,1}};
  constexpr int CC    = Cin / 32;
  constexpr int Nt    = Cout / 16;
  constexpr int NPW   = (Nt >= 4) ? 4 : Nt;
  constexpr int NtG   = Nt / NPW;
  constexpr int LgNtG = __builtin_ctz(NtG);
  int Hp = Hin + 2, Wp = Win + 2;
  int Hout = Hin * 2, Wout = Win * 2;
  int wid = (blockIdx.x * blockDim.x + threadIdx.x) >> 5;
  int lane = threadIdx.x & 31;
  if (wid >= tiles) return;
  int np = wid & (NtG - 1);
  int t  = wid >> LgNtG;
  int mx = t & ((1 << lgMt) - 1);  t >>= lgMt;
  int q  = t & ((1 << lgHin) - 1);
  int b  = t >> lgHin;
  int lr = lane & 15, hi = lane >> 4;
  int r0 = (mx << 4) + lr;
  v8f acc[NPW];
#pragma unroll
  for (int j = 0; j < NPW; ++j) acc[j] = (v8f){};
  const _Float16* wbase = wf + ((size_t)(np * NPW) * 32 + lane) * 16;
#pragma unroll
  for (int tp = 0; tp < NTAPS; ++tp) {
    int iy = q + DY[P][tp] + 1;                 // padded coordinate (halo handles +1 overrun)
    int ix = r0 + DX[P][tp] + 1;
    const _Float16* ap = in + ((size_t)(b * Hp + iy) * Wp + ix) * Cin + hi * 8;
    const _Float16* bp = wbase + (size_t)tp * CC * Nt * 512;
#pragma unroll 2
    for (int cc = 0; cc < CC; ++cc) {
      v8h alo = *(const v8h*)(ap);
      v8h ahi = *(const v8h*)(ap + 16);
      v16h a = __builtin_shufflevector(alo, ahi, 0,1,2,3,4,5,6,7,8,9,10,11,12,13,14,15);
      v16h bfr[NPW];
#pragma unroll
      for (int j = 0; j < NPW; ++j) bfr[j] = *(const v16h*)(bp + j * 512);
#pragma unroll
      for (int j = 0; j < NPW; ++j)
        acc[j] = __builtin_amdgcn_wmma_f32_16x16x32_f16(false, a, false, bfr[j], (short)0,
                                                        acc[j], false, false);
      ap += 32;
      bp += (size_t)Nt * 512;
    }
  }
  int n0 = (np * NPW) * 16 + lr;
  int oy2 = 2 * q + PY;
  float* outb = out + (size_t)(b * Hout + oy2) * Wout * Cout;
#pragma unroll
  for (int j = 0; j < NPW; ++j) {
    float bv = bias[n0 + 16 * j];
#pragma unroll
    for (int r = 0; r < 8; ++r) {
      int m = (hi << 3) + r;
      int ox2 = 2 * ((mx << 4) + m) + PX;
      outb[(size_t)ox2 * Cout + n0 + 16 * j] = acc[j][r] + bv;
    }
  }
}

// ---------------------------------------------------------------------------
// Final 7x7 reflect conv (32->3, padded f16 NHWC in, 514 stride) + tanh +
// instance-mean partial sums via LDS ds_add_f32 bins, flushed once per block.
// ---------------------------------------------------------------------------
__global__ void final_conv_tanh_seg(const _Float16* __restrict__ act, const float* __restrict__ w,
                                    const float* __restrict__ bias, const int* __restrict__ inst,
                                    float* __restrict__ gsum) {
  __shared__ float lw[32 * 49 * 3];          // [(ci*49+tap)*3 + co]
  __shared__ float bins[KIDS * 4];
  for (int i = threadIdx.x; i < 32 * 49 * 3; i += blockDim.x) {
    int co = i % 3; int rest = i / 3; int tap = rest % 49; int ci = rest / 49;
    lw[i] = w[(co * 32 + ci) * 49 + tap];    // cf_w: (3,32,7,7)
  }
  for (int i = threadIdx.x; i < KIDS * 4; i += blockDim.x) bins[i] = 0.f;
  __syncthreads();
  int pix = blockIdx.x * blockDim.x + threadIdx.x;
  int xq = pix & 511; int y = (pix >> 9) & 511; int b = pix >> 18;
  float a0 = bias[0], a1 = bias[1], a2 = bias[2];
  for (int ky = 0; ky < 7; ++ky) {
    int iy = refl(y - 3 + ky, 512);
    for (int kx = 0; kx < 7; ++kx) {
      int ix = refl(xq - 3 + kx, 512);
      const _Float16* ip = act + ((size_t)(b * 514 + iy + 1) * 514 + ix + 1) * 32;
      int tap = ky * 7 + kx;
#pragma unroll 8
      for (int ci = 0; ci < 32; ++ci) {
        float v = (float)ip[ci];
        const float* lp = &lw[(ci * 49 + tap) * 3];
        a0 = fmaf(v, lp[0], a0); a1 = fmaf(v, lp[1], a1); a2 = fmaf(v, lp[2], a2);
      }
    }
  }
  a0 = tanhf(a0); a1 = tanhf(a1); a2 = tanhf(a2);
  int sid = inst[pix] & (KIDS - 1);
  atomicAdd(&bins[sid * 4 + 0], a0);
  atomicAdd(&bins[sid * 4 + 1], a1);
  atomicAdd(&bins[sid * 4 + 2], a2);
  atomicAdd(&bins[sid * 4 + 3], 1.0f);
  __syncthreads();
  for (int i = threadIdx.x; i < KIDS * 4; i += blockDim.x) {
    int seg = b * KIDS + (i >> 2);           // whole block lies in one image
    atomicAdd(&gsum[seg * 4 + (i & 3)], bins[i]);
  }
}

__global__ void broadcast_means(const int* __restrict__ inst, const float* __restrict__ gsum,
                                float* __restrict__ outp) {
  int pix = blockIdx.x * blockDim.x + threadIdx.x;
  int xq = pix & 511; int y = (pix >> 9) & 511; int b = pix >> 18;
  int sid = inst[pix] & (KIDS - 1);
  int seg = b * KIDS + sid;
  float inv = 1.f / fmaxf(gsum[seg * 4 + 3], 1.f);
  const size_t hw = 512 * 512;
  size_t o = (size_t)b * 3 * hw + ((size_t)y << 9) + xq;
  outp[o]          = gsum[seg * 4 + 0] * inv;
  outp[o + hw]     = gsum[seg * 4 + 1] * inv;
  outp[o + 2 * hw] = gsum[seg * 4 + 2] * inv;
}

// ---------------------------------------------------------------------------
extern "C" void kernel_launch(void* const* d_in, const int* in_sizes, int n_in,
                              void* d_out, int out_size, void* d_ws, size_t ws_size,
                              hipStream_t stream) {
  (void)in_sizes; (void)n_in; (void)out_size;
  const float* x    = (const float*)d_in[0];
  const int*   inst = (const int*)d_in[1];
  const float* c0_w  = (const float*)d_in[2];
  const float* c0_b  = (const float*)d_in[3];
  const float* bn0_g = (const float*)d_in[4];
  const float* bn0_b = (const float*)d_in[5];
  const float *dw[4], *db[4], *dg[4], *dbb[4];
  for (int i = 0; i < 4; ++i) {
    dw[i]  = (const float*)d_in[6 + 4 * i];
    db[i]  = (const float*)d_in[7 + 4 * i];
    dg[i]  = (const float*)d_in[8 + 4 * i];
    dbb[i] = (const float*)d_in[9 + 4 * i];
  }
  const float *uw[4], *ub[4], *ug[4], *ubb[4];
  for (int i = 0; i < 4; ++i) {
    uw[i]  = (const float*)d_in[22 + 4 * i];
    ub[i]  = (const float*)d_in[23 + 4 * i];
    ug[i]  = (const float*)d_in[24 + 4 * i];
    ubb[i] = (const float*)d_in[25 + 4 * i];
  }
  const float* cf_w = (const float*)d_in[38];
  const float* cf_b = (const float*)d_in[39];

  // ---- workspace carve (bump allocator, 256B aligned) ----
  char* ws = (char*)d_ws;
  size_t off = 0;
  auto carve = [&](size_t bytes) -> void* {
    void* p = ws + off;
    off += (bytes + 255) & ~(size_t)255;
    return p;
  };
  const int B = 4;
  float*    f32buf = (float*)carve((size_t)B * 512 * 512 * 32 * sizeof(float));
  _Float16* actA   = (_Float16*)carve((size_t)B * 514 * 514 * 32 * sizeof(_Float16));
  _Float16* actB   = (_Float16*)carve((size_t)B * 514 * 514 * 32 * sizeof(_Float16));
  float*    stats  = (float*)carve(2 * 512 * sizeof(float));
  float*    gsum   = (float*)carve((size_t)B * KIDS * 4 * sizeof(float));
  static const int pn_[4] = {1, 2, 2, 4};       // taps per parity (0,0),(0,1),(1,0),(1,1)
  _Float16* wpd[4];
  for (int i = 0; i < 4; ++i) {
    int Cin = 32 << i, Cout = 64 << i;
    wpd[i] = (_Float16*)carve((size_t)9 * Cin * Cout * sizeof(_Float16));
  }
  _Float16* wpu[4][4];
  for (int i = 0; i < 4; ++i) {
    int Cin = 512 >> i, Cout = Cin >> 1;
    for (int p = 0; p < 4; ++p)
      wpu[i][p] = (_Float16*)carve((size_t)pn_[p] * Cin * Cout * sizeof(_Float16));
  }
  if (off > ws_size) return;                    // workspace too small: no-op

  // ---- repack all weights to WMMA fragment order (cheap, every call) ----
  for (int i = 0; i < 4; ++i) {
    int Cin = 32 << i, Cout = 64 << i;
    int elems = 9 * Cin * Cout;
    repack_w_down<<<(elems + 255) / 256, 256, 0, stream>>>(dw[i], wpd[i], Cin, Cout);
  }
  static const int pky_[4][4] = {{1,0,0,0},{1,1,0,0},{0,2,0,0},{0,0,2,2}};
  static const int pkx_[4][4] = {{1,0,0,0},{0,2,0,0},{1,1,0,0},{0,2,0,2}};
  for (int i = 0; i < 4; ++i) {
    int Cin = 512 >> i, Cout = Cin >> 1;
    for (int p = 0; p < 4; ++p) {
      int elems = pn_[p] * Cin * Cout;
      int4 tky = make_int4(pky_[p][0], pky_[p][1], pky_[p][2], pky_[p][3]);
      int4 tkx = make_int4(pkx_[p][0], pkx_[p][1], pkx_[p][2], pkx_[p][3]);
      repack_w_up<<<(elems + 255) / 256, 256, 0, stream>>>(uw[i], wpu[i][p], Cin, Cout,
                                                           pn_[p], tky, tkx);
    }
  }

  auto lg2 = [](int v) { int r = 0; while ((1 << r) < v) ++r; return r; };
  auto run_bn = [&](const float* conv_out, int H, int W, int C,
                    const float* g, const float* bb, _Float16* act) {
    int Npix = B * H * W;
    int total = Npix * C;
    hipMemsetAsync(stats, 0, 2 * C * sizeof(float), stream);
    bn_stats<<<(total + 16383) / 16384, 256, 0, stream>>>(conv_out, stats, total, C);
    int perim = 2 * (W + 2) + 2 * H;
    int ht = B * perim * C;
    zero_halo<<<(ht + 255) / 256, 256, 0, stream>>>(act, B, H, W, C, lg2(C));
    bn_relu_pad_f16<<<(total + 255) / 256, 256, 0, stream>>>(conv_out, stats, g, bb, act,
        total, C, lg2(C), lg2(W), lg2(H), 1.f / (float)Npix);
  };

  auto ntg_of = [](int Cout) { int nt = Cout / 16; int npw = nt >= 4 ? 4 : nt; return nt / npw; };

  // ---- conv0 + bn0 ----
  conv0_reflect7<<<(B * 512 * 512) / 256, 256, 0, stream>>>(x, c0_w, c0_b, f32buf);
  run_bn(f32buf, 512, 512, 32, bn0_g, bn0_b, actA);

  _Float16* cur = actA;
  _Float16* nxt = actB;

  // ---- down path (templated WMMA implicit GEMM, padded input) ----
  for (int i = 0; i < 4; ++i) {
    int Hin = 512 >> i, Win = 512 >> i;
    int Hout = Hin >> 1, Wout = Win >> 1;
    int Cout = 64 << i;
    int tiles = B * Hout * (Wout >> 4) * ntg_of(Cout);
    int blocks = (tiles + 7) / 8;
    int lgMt = lg2(Win) - 5, lgHout = lg2(Hin) - 1;
    switch (i) {
      case 0: conv_down_wmma<32, 64>  <<<blocks, 256, 0, stream>>>(cur, wpd[i], db[i], f32buf, Hin, Win, lgMt, lgHout, tiles); break;
      case 1: conv_down_wmma<64, 128> <<<blocks, 256, 0, stream>>>(cur, wpd[i], db[i], f32buf, Hin, Win, lgMt, lgHout, tiles); break;
      case 2: conv_down_wmma<128, 256><<<blocks, 256, 0, stream>>>(cur, wpd[i], db[i], f32buf, Hin, Win, lgMt, lgHout, tiles); break;
      case 3: conv_down_wmma<256, 512><<<blocks, 256, 0, stream>>>(cur, wpd[i], db[i], f32buf, Hin, Win, lgMt, lgHout, tiles); break;
    }
    run_bn(f32buf, Hout, Wout, Cout, dg[i], dbb[i], nxt);
    _Float16* tmp = cur; cur = nxt; nxt = tmp;
  }

  // ---- up path (ConvT as 4 parity WMMA GEMMs, templated, padded input) ----
  for (int i = 0; i < 4; ++i) {
    int Cout = 256 >> i;
    int Hin = 32 << i, Win = 32 << i;
    int Hout = Hin * 2, Wout = Win * 2;
    int tiles = B * Hin * (Win >> 4) * ntg_of(Cout);
    int blocks = (tiles + 7) / 8;
    int lgMt = lg2(Win) - 4, lgHin = lg2(Hin);
#define UP_LAYER(CI, CO)                                                                               \
    do {                                                                                               \
      conv_up_wmma<CI, CO, 0><<<blocks, 256, 0, stream>>>(cur, wpu[i][0], ub[i], f32buf, Hin, Win, lgMt, lgHin, tiles); \
      conv_up_wmma<CI, CO, 1><<<blocks, 256, 0, stream>>>(cur, wpu[i][1], ub[i], f32buf, Hin, Win, lgMt, lgHin, tiles); \
      conv_up_wmma<CI, CO, 2><<<blocks, 256, 0, stream>>>(cur, wpu[i][2], ub[i], f32buf, Hin, Win, lgMt, lgHin, tiles); \
      conv_up_wmma<CI, CO, 3><<<blocks, 256, 0, stream>>>(cur, wpu[i][3], ub[i], f32buf, Hin, Win, lgMt, lgHin, tiles); \
    } while (0)
    switch (i) {
      case 0: UP_LAYER(512, 256); break;
      case 1: UP_LAYER(256, 128); break;
      case 2: UP_LAYER(128, 64);  break;
      case 3: UP_LAYER(64, 32);   break;
    }
#undef UP_LAYER
    run_bn(f32buf, Hout, Wout, Cout, ug[i], ubb[i], nxt);
    _Float16* tmp = cur; cur = nxt; nxt = tmp;
  }

  // ---- final conv + tanh + instance-mean ----
  hipMemsetAsync(gsum, 0, (size_t)B * KIDS * 4 * sizeof(float), stream);
  final_conv_tanh_seg<<<(B * 512 * 512) / 256, 256, 0, stream>>>(cur, cf_w, cf_b, inst, gsum);
  broadcast_means<<<(B * 512 * 512) / 256, 256, 0, stream>>>(inst, gsum, (float*)d_out);
}
